// AdvancedGraphWaveletTransform_3006477107889
// MI455X (gfx1250) — matrix operand order repaired and checked by reading, not compiled, not run on hardware
//
#include <hip/hip_runtime.h>

// ---------------- problem constants ----------------
#define BB   8
#define NNP  2048
#define CIN  3
#define DDIM 64
#define KNBR 8
#define LVLS 3
#define HID1 256
#define HID2 128
#define NPTS (BB * NNP)          // 16384 points total

// ---------------- vector types ----------------
typedef __attribute__((ext_vector_type(16))) __bf16 v16bf;
typedef __attribute__((ext_vector_type(8)))  float  v8f;
typedef unsigned short ushort8  __attribute__((ext_vector_type(8)));
typedef unsigned short ushort16 __attribute__((ext_vector_type(16)));

// ---------------- workspace layout (bytes) ----------------
static constexpr size_t OFF_FEAT  = 0;                          // bf16 feat   [NPTS,64]   2 MB
static constexpr size_t OFF_FW    = OFF_FEAT  + (size_t)NPTS*64*2;   // f32 fw [NPTS,3]
static constexpr size_t OFF_IDX   = OFF_FW    + (size_t)NPTS*3*4;    // int idx [NPTS,8]
static constexpr size_t OFF_W2T   = OFF_IDX   + (size_t)NPTS*8*4;    // bf16 W2 tiled  128*64
static constexpr size_t OFF_WF1T  = OFF_W2T   + (size_t)128*64*2;    // bf16 Wf1 tiled 192*256
static constexpr size_t OFF_WF2T  = OFF_WF1T  + (size_t)192*256*2;   // bf16 Wf2 tiled 256*128
static constexpr size_t OFF_MULTI = OFF_WF2T  + (size_t)256*128*2;   // bf16 multi [NPTS,192] 6 MB
static constexpr size_t OFF_H1    = OFF_MULTI + (size_t)NPTS*192*2;  // bf16 h1 [NPTS,256]    8 MB
static constexpr size_t OFF_H2    = OFF_H1    + (size_t)NPTS*256*2;  // f32  h2 [NPTS,128]    8 MB
// total ~26 MB of scratch

// ---------------- helpers ----------------
__device__ __forceinline__ unsigned short f2bf(float f) {
  union { float f; unsigned u; } v; v.f = f;
  unsigned r = v.u + 0x7FFFu + ((v.u >> 16) & 1u);   // round-to-nearest-even
  return (unsigned short)(r >> 16);
}
__device__ __forceinline__ float leaky(float v) { return v > 0.f ? v : 0.2f * v; }

__device__ __forceinline__ v16bf mk16(ushort8 lo, ushort8 hi) {
  ushort16 u;
#pragma unroll
  for (int e = 0; e < 8; ++e) { u[e] = lo[e]; u[e + 8] = hi[e]; }
  return __builtin_bit_cast(v16bf, u);
}

// =====================================================================
// 0) Weight prep: f32 [K,Nc] row-major  ->  bf16 tiled B-fragment layout
//    dst[((cb*KT+kb)*32 + lane)*16 + e] = bf16(W[k, col])
//    k = kb*32 + (lane>=16 ? 16 : 0) + e ;  col = cb*16 + (lane&15)
// =====================================================================
__global__ void prep_weight_kernel(const float* __restrict__ W,
                                   unsigned short* __restrict__ Wt,
                                   int K, int Nc, int KT) {
  int tid = blockIdx.x * blockDim.x + threadIdx.x;
  if (tid >= K * Nc) return;
  int e    = tid & 15;
  int lane = (tid >> 4) & 31;
  int t    = tid >> 9;
  int kb   = t % KT;
  int cb   = t / KT;
  int k    = kb * 32 + ((lane & 16) ? 16 : 0) + e;
  int col  = cb * 16 + (lane & 15);
  Wt[tid] = f2bf(W[k * Nc + col]);
}

// =====================================================================
// 1) Per-point feature transform (3->64, BN, leaky) + suppressor MLP
//    (3->64 relu ->3 sigmoid). One 64-thread block per point.
// =====================================================================
__global__ void feat_fw_kernel(const float* __restrict__ x,
                               const float* __restrict__ W1, const float* __restrict__ b1,
                               const float* __restrict__ g1, const float* __restrict__ be1,
                               const float* __restrict__ Ws1, const float* __restrict__ bs1,
                               const float* __restrict__ Ws2, const float* __restrict__ bs2,
                               unsigned short* __restrict__ featbf,
                               float* __restrict__ fw) {
  const int p = blockIdx.x;       // global point id
  const int d = threadIdx.x;      // 0..63
  const float x0 = x[p * 3 + 0], x1 = x[p * 3 + 1], x2 = x[p * 3 + 2];

  float f = fmaf(x0, W1[d], fmaf(x1, W1[64 + d], fmaf(x2, W1[128 + d], b1[d])));
  f = leaky(f * g1[d] + be1[d]);
  featbf[p * 64 + d] = f2bf(f);

  float h = fmaf(x0, Ws1[d], fmaf(x1, Ws1[64 + d], fmaf(x2, Ws1[128 + d], bs1[d])));
  h = h > 0.f ? h : 0.f;
  __shared__ float hid[64];
  hid[d] = h;
  __syncthreads();
  if (d < LVLS) {
    float acc = bs2[d];
    for (int j = 0; j < 64; ++j) acc = fmaf(hid[j], Ws2[j * 3 + d], acc);
    fw[p * 3 + d] = 1.f / (1.f + __expf(-acc));
  }
}

// =====================================================================
// 2) KNN top-8 (squared distance, includes self just like top_k(-d2)).
//    One thread per query point; LDS tiles of 256 candidate points.
// =====================================================================
__global__ void __launch_bounds__(256) knn_kernel(const float* __restrict__ x,
                                                  int* __restrict__ idxg) {
  const int tid = threadIdx.x;
  const int b   = blockIdx.x >> 3;                 // 8 blocks per batch
  const int n   = ((blockIdx.x & 7) << 8) + tid;
  const int gp  = b * NNP + n;
  const float qx = x[gp * 3], qy = x[gp * 3 + 1], qz = x[gp * 3 + 2];

  float bd[8]; int bi[8];
#pragma unroll
  for (int s = 0; s < 8; ++s) { bd[s] = 3.4e38f; bi[s] = gp; }

  __shared__ float sx[256], sy[256], sz[256];
  for (int t = 0; t < NNP / 256; ++t) {
    int gm = b * NNP + t * 256 + tid;
    sx[tid] = x[gm * 3]; sy[tid] = x[gm * 3 + 1]; sz[tid] = x[gm * 3 + 2];
    __syncthreads();
    for (int j = 0; j < 256; ++j) {
      float dx = qx - sx[j], dy = qy - sy[j], dz = qz - sz[j];
      float d2 = fmaf(dx, dx, fmaf(dy, dy, dz * dz));
      // find current worst of the 8
      float wm = bd[0]; int wsl = 0;
#pragma unroll
      for (int s = 1; s < 8; ++s) { if (bd[s] > wm) { wm = bd[s]; wsl = s; } }
      if (d2 < wm) {
        int gm2 = b * NNP + t * 256 + j;
#pragma unroll
        for (int s = 0; s < 8; ++s) { if (s == wsl) { bd[s] = d2; bi[s] = gm2; } }
      }
    }
    __syncthreads();
  }
#pragma unroll
  for (int s = 0; s < 8; ++s) idxg[gp * 8 + s] = bi[s];
}

// =====================================================================
// 3) Edge conv via WMMA bf16: rows = (point,kk), K = 128 (center||nbr),
//    cols = 64. One wave = 16 rows = 2 points * 8 neighbors.
//    Epilogue fuses bias->BN->leaky->max_k->fw-level-scale, writes multi.
// =====================================================================
__global__ void __launch_bounds__(128) edgeconv_kernel(
    const unsigned short* __restrict__ featbf,
    const int* __restrict__ idxg,
    const unsigned short* __restrict__ W2t,
    const float* __restrict__ b2, const float* __restrict__ g2,
    const float* __restrict__ be2,
    const float* __restrict__ fw,
    unsigned short* __restrict__ multibf) {
  const int lane  = threadIdx.x & 31;
  const int w     = threadIdx.x >> 5;
  const int pbase = blockIdx.x * 8 + w * 2;      // 2 points per wave
  const int M     = lane & 15;                   // A-row this lane feeds
  const int pm    = pbase + (M >> 3);
  const int kk    = M & 7;
  const int nb    = idxg[pm * 8 + kk];           // global neighbor point

  const unsigned short* cptr = featbf + (size_t)pm * 64;
  const unsigned short* nptr = featbf + (size_t)nb * 64;

  // A fragments: kb 0..1 -> center features, kb 2..3 -> neighbor features
  v16bf afrag[4];
#pragma unroll
  for (int kb = 0; kb < 4; ++kb) {
    const unsigned short* rp = (kb < 2) ? cptr : nptr;
    int koff = ((kb & 1) << 5) + ((lane & 16) ? 16 : 0);
    afrag[kb] = mk16(*(const ushort8*)(rp + koff),
                     *(const ushort8*)(rp + koff + 8));
  }

  const int point = pbase + ((lane & 16) ? 1 : 0);
  const float fw0 = fw[point * 3 + 0];
  const float fw1 = fw[point * 3 + 1];
  const float fw2 = fw[point * 3 + 2];

  __builtin_prefetch(W2t, 0, 0);                 // global_prefetch_b8

#pragma unroll
  for (int cb = 0; cb < 4; ++cb) {
    v8f acc = {0.f, 0.f, 0.f, 0.f, 0.f, 0.f, 0.f, 0.f};
#pragma unroll
    for (int kb = 0; kb < 4; ++kb) {
      const unsigned short* p = W2t + (size_t)(((cb << 2) + kb) * 32 + lane) * 16;
      v16bf bfrag = mk16(*(const ushort8*)p, *(const ushort8*)(p + 8));
      acc = __builtin_amdgcn_wmma_f32_16x16x32_bf16(false, afrag[kb], false, bfrag,
                                                    (short)0, acc, false, false);
    }
    const int col = (cb << 4) + (lane & 15);
    const float bb = b2[col], gg = g2[col], ee = be2[col];
    float m = -3.4e38f;
#pragma unroll
    for (int j = 0; j < 8; ++j) {                // max over the 8 k-rows
      float v = leaky((acc[j] + bb) * gg + ee);
      m = fmaxf(m, v);
    }
    unsigned short* mb = multibf + (size_t)point * (LVLS * DDIM) + col;
    mb[0]   = f2bf(m * fw0);
    mb[64]  = f2bf(m * fw1);
    mb[128] = f2bf(m * fw2);
  }
}

// =====================================================================
// 4) Fusion GEMM (bf16 WMMA, f32 acc): A[M,K] @ W[K,NC], bias->BN->leaky.
//    STOREF32 picks bf16 (feeds next GEMM) vs f32 (feeds final proj).
// =====================================================================
template <int KT, int NC, bool STOREF32>
__global__ void __launch_bounds__(128) fusion_gemm_kernel(
    const unsigned short* __restrict__ A,
    const unsigned short* __restrict__ Wt,
    const float* __restrict__ bias, const float* __restrict__ gam,
    const float* __restrict__ bet,
    unsigned short* __restrict__ outbf, float* __restrict__ outf) {
  const int lane   = threadIdx.x & 31;
  const int w      = threadIdx.x >> 5;
  const int tile_m = (blockIdx.x * 4 + w) * 16;
  const int K      = KT * 32;
  const int arow   = tile_m + (lane & 15);
  const unsigned short* rp = A + (size_t)arow * K;

  v16bf afrag[KT];
#pragma unroll
  for (int kb = 0; kb < KT; ++kb) {
    int koff = kb * 32 + ((lane & 16) ? 16 : 0);
    afrag[kb] = mk16(*(const ushort8*)(rp + koff),
                     *(const ushort8*)(rp + koff + 8));
  }

  __builtin_prefetch(Wt, 0, 0);

#pragma unroll
  for (int cb = 0; cb < NC / 16; ++cb) {
    v8f acc = {0.f, 0.f, 0.f, 0.f, 0.f, 0.f, 0.f, 0.f};
#pragma unroll
    for (int kb = 0; kb < KT; ++kb) {
      const unsigned short* p = Wt + (size_t)((cb * KT + kb) * 32 + lane) * 16;
      v16bf bfrag = mk16(*(const ushort8*)p, *(const ushort8*)(p + 8));
      acc = __builtin_amdgcn_wmma_f32_16x16x32_bf16(false, afrag[kb], false, bfrag,
                                                    (short)0, acc, false, false);
    }
    const int col = cb * 16 + (lane & 15);
    const float bb = bias[col], gg = gam[col], ee = bet[col];
#pragma unroll
    for (int j = 0; j < 8; ++j) {
      int r = tile_m + j + ((lane & 16) ? 8 : 0);
      float v = leaky((acc[j] + bb) * gg + ee);
      if (STOREF32) outf[(size_t)r * NC + col] = v;
      else          outbf[(size_t)r * NC + col] = f2bf(v);
    }
  }
}

// =====================================================================
// 5) Final projection 128->3 (f32) + residual:  out = x + 0.1*delta
// =====================================================================
__global__ void final_kernel(const float* __restrict__ h2,
                             const float* __restrict__ Wf3,
                             const float* __restrict__ bf3,
                             const float* __restrict__ x,
                             float* __restrict__ out) {
  int tid = blockIdx.x * blockDim.x + threadIdx.x;
  if (tid >= NPTS * 3) return;
  int p = tid / 3, c = tid % 3;
  float acc = bf3[c];
  const float* hr = h2 + (size_t)p * HID2;
#pragma unroll 8
  for (int k = 0; k < HID2; ++k) acc = fmaf(hr[k], Wf3[k * 3 + c], acc);
  out[tid] = x[tid] + 0.1f * acc;
}

// =====================================================================
// launcher
// =====================================================================
extern "C" void kernel_launch(void* const* d_in, const int* in_sizes, int n_in,
                              void* d_out, int out_size, void* d_ws, size_t ws_size,
                              hipStream_t stream) {
  (void)in_sizes; (void)n_in; (void)out_size; (void)ws_size;
  const float* x    = (const float*)d_in[0];
  const float* W1   = (const float*)d_in[1];
  const float* b1   = (const float*)d_in[2];
  const float* g1   = (const float*)d_in[3];
  const float* be1  = (const float*)d_in[4];
  const float* W2   = (const float*)d_in[5];
  const float* b2   = (const float*)d_in[6];
  const float* g2   = (const float*)d_in[7];
  const float* be2  = (const float*)d_in[8];
  const float* Ws1  = (const float*)d_in[9];
  const float* bs1  = (const float*)d_in[10];
  const float* Ws2  = (const float*)d_in[11];
  const float* bs2  = (const float*)d_in[12];
  const float* Wf1  = (const float*)d_in[13];
  const float* bf1  = (const float*)d_in[14];
  const float* gf1  = (const float*)d_in[15];
  const float* bef1 = (const float*)d_in[16];
  const float* Wf2  = (const float*)d_in[17];
  const float* bf2  = (const float*)d_in[18];
  const float* gf2  = (const float*)d_in[19];
  const float* bef2 = (const float*)d_in[20];
  const float* Wf3  = (const float*)d_in[21];
  const float* bf3  = (const float*)d_in[22];
  float* out = (float*)d_out;

  char* ws = (char*)d_ws;
  unsigned short* featbf  = (unsigned short*)(ws + OFF_FEAT);
  float*          fw      = (float*)(ws + OFF_FW);
  int*            idxg    = (int*)(ws + OFF_IDX);
  unsigned short* W2t     = (unsigned short*)(ws + OFF_W2T);
  unsigned short* Wf1t    = (unsigned short*)(ws + OFF_WF1T);
  unsigned short* Wf2t    = (unsigned short*)(ws + OFF_WF2T);
  unsigned short* multibf = (unsigned short*)(ws + OFF_MULTI);
  unsigned short* h1bf    = (unsigned short*)(ws + OFF_H1);
  float*          h2f     = (float*)(ws + OFF_H2);

  // weight pre-swizzle into WMMA B-fragment layout (bf16)
  prep_weight_kernel<<<(128 * 64 + 255) / 256, 256, 0, stream>>>(W2, W2t, 128, 64, 4);
  prep_weight_kernel<<<(192 * 256 + 255) / 256, 256, 0, stream>>>(Wf1, Wf1t, 192, 256, 6);
  prep_weight_kernel<<<(256 * 128 + 255) / 256, 256, 0, stream>>>(Wf2, Wf2t, 256, 128, 8);

  // per-point features + suppressor weights
  feat_fw_kernel<<<NPTS, 64, 0, stream>>>(x, W1, b1, g1, be1, Ws1, bs1, Ws2, bs2,
                                          featbf, fw);

  // KNN (top-8 by squared distance, self included)
  knn_kernel<<<NPTS / 256, 256, 0, stream>>>(x, idxg);

  // edge conv GEMM (WMMA) with fused max_k + fw scaling -> multi[NPTS,192]
  edgeconv_kernel<<<NPTS / 8, 128, 0, stream>>>(featbf, idxg, W2t, b2, g2, be2,
                                                fw, multibf);

  // fusion MLP (WMMA): 192->256 (bf16 out), 256->128 (f32 out)
  fusion_gemm_kernel<6, 256, false><<<NPTS / 64, 128, 0, stream>>>(
      multibf, Wf1t, bf1, gf1, bef1, h1bf, nullptr);
  fusion_gemm_kernel<8, 128, true><<<NPTS / 64, 128, 0, stream>>>(
      h1bf, Wf2t, bf2, gf2, bef2, nullptr, h2f);

  // final 128->3 + residual
  final_kernel<<<(NPTS * 3 + 255) / 256, 256, 0, stream>>>(h2f, Wf3, bf3, x, out);
}
